// QLSTM_65481071405411
// MI455X (gfx1250) — matrix operand order repaired and checked by reading, not compiled, not run on hardware
//
#include <hip/hip_runtime.h>
#include <hip/hip_bf16.h>

// ---------------------------------------------------------------------------
// QLSTM collapsed to closed form (see analysis):
//   angle[g] = Weff_g · [x, hx] + beff_g                  (Weff = W2·W1 folded)
//   z_w      = a_w + b_w·cos(angle_w) + d_w·sin(angle_w)  (per-wire Z expectation)
//   gate_q   = prefix-product_{w<=q} z_w                  (CNOT chain == prefix XOR)
//   standard LSTM combine.
// x-part of the angle GEMM for all T*B rows is one big WMMA GEMM:
//   (262144 x 64) * (64 x 32) -> preX, f16 inputs / f32 accumulate.
// ---------------------------------------------------------------------------

typedef __attribute__((ext_vector_type(16))) _Float16 v16h;
typedef __attribute__((ext_vector_type(8)))  float    v8f;

#define Tn 256
#define Bn 1024
#define DIN 64
#define NQ 8
#define INF 72   // DIN + NQ

struct GateParams {
    const float* W1[4];
    const float* b1[4];
    const float* W2[4];
    const float* b2[4];
    const float* P[4];
};

// ---------------------------------------------------------------------------
// Kernel 1: fold weights, derive per-wire z-coefficients from Rot params.
// ---------------------------------------------------------------------------
__global__ void qlstm_precompute_kernel(GateParams p, float* __restrict__ Weff,
                                        float* __restrict__ beff,
                                        float* __restrict__ coeffs) {
    const int tid = threadIdx.x;

    // Weff[n][j], n = g*8+w in 0..31, j in 0..71:  Weff = W2 @ W1
    for (int e = tid; e < 32 * INF; e += blockDim.x) {
        const int n = e / INF, j = e % INF;
        const int g = n >> 3, w = n & 7;
        float s = 0.f;
#pragma unroll
        for (int k = 0; k < 8; ++k)
            s = fmaf(p.W2[g][w * 8 + k], p.W1[g][k * INF + j], s);
        Weff[e] = s;
    }

    if (tid < 32) {
        const int g = tid >> 3, w = tid & 7;

        // beff = W2 @ b1 + b2
        float s = p.b2[g][w];
#pragma unroll
        for (int k = 0; k < 8; ++k) s = fmaf(p.W2[g][w * 8 + k], p.b1[g][k], s);
        beff[tid] = s;

        // Build V = Rot(l=1) * Rot(l=0) for this wire (2x2 complex).
        float Rr[2][2][2], Ri[2][2][2];   // [layer][i][j]
#pragma unroll
        for (int l = 0; l < 2; ++l) {
            const float* q = p.P[g] + l * 24 + w * 3;
            const float phi = q[0], th = q[1], om = q[2];
            const float ct = cosf(0.5f * th), st = sinf(0.5f * th);
            const float a = 0.5f * (phi + om), b = 0.5f * (phi - om);
            const float ca = cosf(a), sa = sinf(a);
            const float cb = cosf(b), sb = sinf(b);
            // m00 = e^{-ia}ct, m01 = -e^{ib}st, m10 = e^{-ib}st, m11 = e^{ia}ct
            Rr[l][0][0] =  ct * ca;  Ri[l][0][0] = -ct * sa;
            Rr[l][0][1] = -st * cb;  Ri[l][0][1] = -st * sb;
            Rr[l][1][0] =  st * cb;  Ri[l][1][0] = -st * sb;
            Rr[l][1][1] =  ct * ca;  Ri[l][1][1] =  ct * sa;
        }
        float Vr[2][2], Vi[2][2];
#pragma unroll
        for (int i = 0; i < 2; ++i)
#pragma unroll
            for (int j = 0; j < 2; ++j) {
                float cr = 0.f, ci = 0.f;
#pragma unroll
                for (int k = 0; k < 2; ++k) {
                    cr += Rr[1][i][k] * Rr[0][k][j] - Ri[1][i][k] * Ri[0][k][j];
                    ci += Rr[1][i][k] * Ri[0][k][j] + Ri[1][i][k] * Rr[0][k][j];
                }
                Vr[i][j] = cr; Vi[i][j] = ci;
            }
        // z = cA + cB*cos(theta) + cD*sin(theta)
        const float A  = (Vr[0][0]*Vr[0][0] + Vi[0][0]*Vi[0][0]) -
                         (Vr[1][0]*Vr[1][0] + Vi[1][0]*Vi[1][0]);
        const float Bc = (Vr[0][1]*Vr[0][1] + Vi[0][1]*Vi[0][1]) -
                         (Vr[1][1]*Vr[1][1] + Vi[1][1]*Vi[1][1]);
        // Im(alpha*conj(beta)) = ai*br - ar*bi
        const float im0 = Vi[0][0]*Vr[0][1] - Vr[0][0]*Vi[0][1];
        const float im1 = Vi[1][0]*Vr[1][1] - Vr[1][0]*Vi[1][1];
        coeffs[tid * 3 + 0] = 0.5f * (A + Bc);
        coeffs[tid * 3 + 1] = 0.5f * (A - Bc);
        coeffs[tid * 3 + 2] = -(im0 - im1);
    }
}

// ---------------------------------------------------------------------------
// Kernel 2: WMMA GEMM  preX[T*B, 32] = X[T*B, 64] * WeffX^T[64, 32] + beff
// One wave computes a 16x32 output tile (2 N-tiles x 2 K-chunks of
// v_wmma_f32_16x16x32_f16).  Fragment packing follows ISA 7.12.2:
//   A (16x32 f16): lane = {half=lane>>4, row=lane&15},
//                  elem i -> K = ((i&8)<<1) + (half<<3) + (i&7)
//   B (32x16 f16): lane = column, elem i -> K = ((i&8)<<1) + (half<<3) + (i&7)
//   C/D (16x16 f32): vgpr r -> M = r + half*8, N = lane&15
// ---------------------------------------------------------------------------
__global__ void qlstm_wmma_gemm_kernel(const float* __restrict__ X,
                                       const float* __restrict__ Weff,
                                       const float* __restrict__ beff,
                                       float* __restrict__ preX) {
    const int lane = threadIdx.x & 31;
    const int wave = (blockIdx.x * blockDim.x + threadIdx.x) >> 5;
    const int half = lane >> 4;
    const int l16  = lane & 15;

    const long row0 = (long)wave * 16;
    const float* xrow = X + (row0 + l16) * DIN;

    v16h a0, a1;
#pragma unroll
    for (int i = 0; i < 16; ++i) {
        const int k = ((i & 8) << 1) + (half << 3) + (i & 7);
        a0[i] = (_Float16)xrow[k];
        a1[i] = (_Float16)xrow[32 + k];
    }

#pragma unroll
    for (int nt = 0; nt < 2; ++nt) {
        const int n = nt * 16 + l16;
        const float* wrow = Weff + n * INF;   // x-part: cols 0..63
        v16h b0, b1;
#pragma unroll
        for (int i = 0; i < 16; ++i) {
            const int k = ((i & 8) << 1) + (half << 3) + (i & 7);
            b0[i] = (_Float16)wrow[k];
            b1[i] = (_Float16)wrow[32 + k];
        }
        v8f c = {};
        c = __builtin_amdgcn_wmma_f32_16x16x32_f16(false, a0, false, b0,
                                                   (short)0, c, false, false);
        c = __builtin_amdgcn_wmma_f32_16x16x32_f16(false, a1, false, b1,
                                                   (short)0, c, false, false);
        const float bias = beff[n];
        float* outp = preX + row0 * 32 + nt * 16 + l16;
#pragma unroll
        for (int r = 0; r < 8; ++r) {
            const int m = r + (half << 3);
            outp[(long)m * 32] = c[r] + bias;
        }
    }
}

// ---------------------------------------------------------------------------
// Kernel 3: sequential LSTM scan.  One wave per batch element;
// lane = gate*8 + wire.  Cross-wire prefix product and cross-gate combine
// via wave32 shuffles.
// ---------------------------------------------------------------------------
__global__ void qlstm_scan_kernel(const float* __restrict__ preX,
                                  const float* __restrict__ Weff,
                                  const float* __restrict__ coeffs,
                                  float* __restrict__ out) {
    const int lane  = threadIdx.x & 31;
    const int batch = blockIdx.x * (blockDim.x >> 5) + (threadIdx.x >> 5);
    const int g = lane >> 3;
    const int w = lane & 7;

    float whh[8];
#pragma unroll
    for (int j = 0; j < 8; ++j) whh[j] = Weff[lane * INF + DIN + j];
    const float cA = coeffs[lane * 3 + 0];
    const float cB = coeffs[lane * 3 + 1];
    const float cD = coeffs[lane * 3 + 2];

    float hx[8];
#pragma unroll
    for (int j = 0; j < 8; ++j) hx[j] = 0.f;
    float cxw = 0.f;

    float* hxout = out + (long)Tn * Bn * NQ;
    float* cxout = hxout + (long)Bn * NQ;

    for (int t = 0; t < Tn; ++t) {
        float ang = preX[((long)t * Bn + batch) * 32 + lane];
#pragma unroll
        for (int j = 0; j < 8; ++j) ang = fmaf(whh[j], hx[j], ang);

        float z = fmaf(cB, __cosf(ang), fmaf(cD, __sinf(ang), cA));

        // inclusive prefix product over wires within each gate's 8 lanes
#pragma unroll
        for (int off = 1; off <= 4; off <<= 1) {
            const float other = __shfl(z, lane - off, 32);
            if (w >= off) z *= other;
        }

        // gate nonlinearity (gate 2 == "g" uses tanh, others sigmoid)
        const float act = (g == 2) ? tanhf(z) : 1.f / (1.f + __expf(-z));

        const float fq = __shfl(act, w, 32);
        const float iq = __shfl(act, 8 + w, 32);
        const float gq = __shfl(act, 16 + w, 32);
        const float oq = __shfl(act, 24 + w, 32);

        cxw = fmaf(fq, cxw, iq * gq);
        const float hxw = oq * tanhf(cxw);

#pragma unroll
        for (int j = 0; j < 8; ++j) hx[j] = __shfl(hxw, j, 32);

        if (lane < 8) out[((long)t * Bn + batch) * NQ + w] = hxw;
    }
    if (lane < 8) {
        hxout[batch * NQ + w] = hx[w];
        cxout[batch * NQ + w] = cxw;
    }
}

// ---------------------------------------------------------------------------
extern "C" void kernel_launch(void* const* d_in, const int* in_sizes, int n_in,
                              void* d_out, int out_size, void* d_ws, size_t ws_size,
                              hipStream_t stream) {
    (void)in_sizes; (void)n_in; (void)out_size; (void)ws_size;
    const float* X = (const float*)d_in[0];

    GateParams p;
    for (int g = 0; g < 4; ++g) {
        p.W1[g] = (const float*)d_in[1 + 5 * g];
        p.b1[g] = (const float*)d_in[2 + 5 * g];
        p.W2[g] = (const float*)d_in[3 + 5 * g];
        p.b2[g] = (const float*)d_in[4 + 5 * g];
        p.P[g]  = (const float*)d_in[5 + 5 * g];
    }

    float* ws     = (float*)d_ws;
    float* Weff   = ws;              // 32 x 72
    float* beff   = ws + 2304;       // 32
    float* coeffs = ws + 2336;       // 32 x 3
    float* preX   = ws + 2560;       // T*B x 32  (32 MB)
    float* out    = (float*)d_out;

    qlstm_precompute_kernel<<<1, 256, 0, stream>>>(p, Weff, beff, coeffs);

    // 262144 rows / 16 rows-per-wave = 16384 waves; 8 waves per 256-thr block
    qlstm_wmma_gemm_kernel<<<2048, 256, 0, stream>>>(X, Weff, beff, preX);

    // 1024 batch elements, one wave each -> 128 blocks of 8 waves
    qlstm_scan_kernel<<<128, 256, 0, stream>>>(preX, Weff, coeffs, out);
}
